// GMLVQ_59322088292919
// MI455X (gfx1250) — matrix operand order repaired and checked by reading, not compiled, not run on hardware
//
#include <hip/hip_runtime.h>

typedef __attribute__((ext_vector_type(16))) __bf16 v16bf;
typedef __attribute__((ext_vector_type(8)))  __bf16 v8bf;
typedef __attribute__((ext_vector_type(8)))  float  v8f;
typedef __attribute__((ext_vector_type(4)))  float  v4f;

#define EPS_R 1e-6f

// ---------------------------------------------------------------------------
// Prep: Wh/Wl = bf16 hi/lo split of (W * rc), w_sq[c] = sum_f rc*W^2, rc out.
// Wl is stored contiguously after Wh (single base for the main kernel's
// immediate-offset addressing). One block per class.
// ---------------------------------------------------------------------------
__global__ void __launch_bounds__(256)
gmlvq_prep(const float* __restrict__ W, const float* __restrict__ r,
           __bf16* __restrict__ Whl,    // [2][C][F]: hi plane then lo plane
           float* __restrict__ wsq, float* __restrict__ rc_out,
           int F, int C) {
  __shared__ float red[256];
  const int c = blockIdx.x;
  const size_t loPlane = (size_t)C * F;
  float acc = 0.f;
  for (int f = threadIdx.x; f < F; f += 256) {
    float rv  = r[f];
    float rcv = rv < EPS_R ? EPS_R : rv;
    float w   = W[(size_t)c * F + f];
    float wr  = w * rcv;
    __bf16 h  = (__bf16)wr;                 // RNE
    __bf16 l  = (__bf16)(wr - (float)h);    // residual
    Whl[(size_t)c * F + f]           = h;
    Whl[loPlane + (size_t)c * F + f] = l;
    acc = fmaf(rcv * w, w, acc);
    if (c == 0) rc_out[f] = rcv;
  }
  red[threadIdx.x] = acc;
  __syncthreads();
  for (int s = 128; s > 0; s >>= 1) {
    if (threadIdx.x < s) red[threadIdx.x] += red[threadIdx.x + s];
    __syncthreads();
  }
  if (threadIdx.x == 0) wsq[c] = red[0];
}

// ---------------------------------------------------------------------------
// Main: per wave, 32 rows x 64 classes. bf16 split-GEMM on
// v_wmma_f32_16x16x32_bf16 (hi*hi + hi*lo + lo*hi into one f32 accumulator),
// exact-f32 x_sq in the WMMA co-execution shadow.
//
// Addressing: one uniform base per array + one 32-bit per-lane byte offset,
// all tile/plane deltas folded into the 24-bit instruction offset (GVS form).
// ---------------------------------------------------------------------------
__global__ void __launch_bounds__(256)
gmlvq_main(const float* __restrict__ X,
           const __bf16* __restrict__ Whl,   // hi plane, lo plane at +C*F
           const float* __restrict__ wsq, const float* __restrict__ rc,
           float* __restrict__ dists, int N, int F) {
  constexpr int C = 64;
  const int tid    = threadIdx.x;
  const int wave   = tid >> 5;
  const int lane   = tid & 31;
  const int laneLo = lane & 15;
  const int laneHi = lane >> 4;          // 0 or 1

  const int rowBase = blockIdx.x * 256 + wave * 32;
  if (rowBase >= N) return;              // uniform per wave

  v8f acc[2][4] = {};                    // [m-tile][c-tile] f32 accumulators

  float wsq_v[4];
#pragma unroll
  for (int ct = 0; ct < 4; ++ct) wsq_v[ct] = wsq[ct * 16 + laneLo];

  // Compile-time-constant deltas (bytes), all < 8MB -> instruction offsets.
  const unsigned dMrow = (unsigned)(16 * F * 4);   // A: m=1 row tile
  const unsigned dCt   = (unsigned)(16 * F * 2);   // B: next c-tile
  const unsigned dLo   = (unsigned)(C * F * 2);    // B: lo plane

  // Per-lane running byte offsets (single 32-bit add per array per chunk).
  unsigned xoff = (unsigned)(((rowBase + laneLo) * F + laneHi * 8) * 4);
  unsigned boff = (unsigned)((laneLo * F + laneHi * 8) * 2);
  unsigned roff = (unsigned)((laneHi * 8) * 4);

  const char* Xb = (const char*)X;
  const char* Bb = (const char*)Whl;
  const char* Rb = (const char*)rc;

  float xsq[2] = {0.f, 0.f};

  for (int k0 = 0; k0 < F; k0 += 32) {
    // rc chunk (matches A element order)
    float rr[16];
    {
      const float* rp = (const float*)(Rb + roff);
      *(v4f*)(rr + 0)  = *(const v4f*)(rp + 0);
      *(v4f*)(rr + 4)  = *(const v4f*)(rp + 4);
      *(v4f*)(rr + 8)  = *(const v4f*)(rp + 16);
      *(v4f*)(rr + 12) = *(const v4f*)(rp + 20);
    }

    // B tiles: 2x8 bf16 per lane (k 8h..8h+7, 16+8h..); hi and lo planes share one base
    // and one VGPR offset, tile/plane selected by immediate offset.
    v16bf Bh[4], Bl[4];
#pragma unroll
    for (int ct = 0; ct < 4; ++ct) {
      const __bf16* ph = (const __bf16*)(Bb + boff + ct * dCt);
      v8bf h0 = *(const v8bf*)(ph);
      v8bf h1 = *(const v8bf*)(ph + 16);
      Bh[ct] = __builtin_shufflevector(h0, h1, 0,1,2,3,4,5,6,7,8,9,10,11,12,13,14,15);
      const __bf16* pl = (const __bf16*)(Bb + boff + ct * dCt + dLo);
      v8bf l0 = *(const v8bf*)(pl);
      v8bf l1 = *(const v8bf*)(pl + 16);
      Bl[ct] = __builtin_shufflevector(l0, l1, 0,1,2,3,4,5,6,7,8,9,10,11,12,13,14,15);
    }

#pragma unroll
    for (int m = 0; m < 2; ++m) {
      // A chunk: 16 f32 per lane in the ISA element order
      const float* xp = (const float*)(Xb + xoff + m * dMrow);
      float xa[16];
      *(v4f*)(xa + 0)  = *(const v4f*)(xp + 0);
      *(v4f*)(xa + 4)  = *(const v4f*)(xp + 4);
      *(v4f*)(xa + 8)  = *(const v4f*)(xp + 16);
      *(v4f*)(xa + 12) = *(const v4f*)(xp + 20);

      if (k0 + 32 < F) __builtin_prefetch(xp + 32, 0, 3);

      v16bf Ah, Al;
#pragma unroll
      for (int e = 0; e < 16; ++e) {
        float x = xa[e];
        xsq[m] = fmaf(rr[e] * x, x, xsq[m]);   // exact f32 diagonal term
        __bf16 h = (__bf16)x;
        Ah[e] = h;
        Al[e] = (__bf16)(x - (float)h);
      }

#pragma unroll
      for (int ct = 0; ct < 4; ++ct) {
        acc[m][ct] = __builtin_amdgcn_wmma_f32_16x16x32_bf16(
            false, Ah, false, Bh[ct], (short)0, acc[m][ct], false, false);
        acc[m][ct] = __builtin_amdgcn_wmma_f32_16x16x32_bf16(
            false, Al, false, Bh[ct], (short)0, acc[m][ct], false, false);
        acc[m][ct] = __builtin_amdgcn_wmma_f32_16x16x32_bf16(
            false, Ah, false, Bl[ct], (short)0, acc[m][ct], false, false);
      }
    }

    xoff += 32 * 4;   // advance one K-chunk
    boff += 32 * 2;
    roff += 32 * 4;
  }

  // Combine the two K-halves of x_sq (lane L and L^16 hold the same row)
#pragma unroll
  for (int m = 0; m < 2; ++m) xsq[m] += __shfl_xor(xsq[m], 16, 32);

  // D layout: VGPR j -> row (m*16 + laneHi*8 + j), col = ct*16 + laneLo
#pragma unroll
  for (int m = 0; m < 2; ++m) {
    const int rbase = rowBase + m * 16 + laneHi * 8;
#pragma unroll
    for (int j = 0; j < 8; ++j) {
      float xs = __shfl(xsq[m], j + laneHi * 8, 32);
      const size_t row = (size_t)(rbase + j);
#pragma unroll
      for (int ct = 0; ct < 4; ++ct) {
        float v = xs + wsq_v[ct] - 2.0f * acc[m][ct][j];
        dists[row * C + ct * 16 + laneLo] = v;
      }
    }
  }
}

// ---------------------------------------------------------------------------
// Argmin over 64 classes per row (first-min tie rule, like jnp.argmin).
// dists is L2-resident (8 MB); this pass is ~free.
// ---------------------------------------------------------------------------
__global__ void __launch_bounds__(256)
gmlvq_argmin(const float* __restrict__ dists, float* __restrict__ preds,
             int N, int C) {
  int n = blockIdx.x * blockDim.x + threadIdx.x;
  if (n >= N) return;
  const float* d = dists + (size_t)n * C;
  float best = d[0];
  int   bi   = 0;
  for (int c = 1; c < C; ++c) {
    float v = d[c];
    if (v < best) { best = v; bi = c; }
  }
  preds[n] = (float)bi;
}

extern "C" void kernel_launch(void* const* d_in, const int* in_sizes, int n_in,
                              void* d_out, int out_size, void* d_ws, size_t ws_size,
                              hipStream_t stream) {
  const float* X = (const float*)d_in[0];
  const float* W = (const float*)d_in[1];
  const float* r = (const float*)d_in[2];

  const int F = in_sizes[2];           // 2048
  const int C = in_sizes[1] / F;       // 64
  const int N = in_sizes[0] / F;       // 32768

  char* ws = (char*)d_ws;
  __bf16* Whl = (__bf16*)ws;                                     // 2*C*F*2 B (hi,lo planes)
  float*  wsq = (float*) (ws + (size_t)C * F * 4);               // 256 B
  float*  rc  = (float*) (ws + (size_t)C * F * 4 + 256);         // F*4 B (16B aligned)

  float* dists = (float*)d_out;
  float* preds = dists + (size_t)N * C;

  gmlvq_prep<<<C, 256, 0, stream>>>(W, r, Whl, wsq, rc, F, C);
  gmlvq_main<<<(N + 255) / 256, 256, 0, stream>>>(X, Whl, wsq, rc, dists, N, F);
  gmlvq_argmin<<<(N + 255) / 256, 256, 0, stream>>>(dists, preds, N, C);
}
